// SparseOverlayLinear_26096221290990
// MI455X (gfx1250) — matrix operand order repaired and checked by reading, not compiled, not run on hardware
//
#include <hip/hip_runtime.h>

typedef __attribute__((ext_vector_type(16))) _Float16 v16h;
typedef __attribute__((ext_vector_type(8)))  _Float16 v8h;
typedef __attribute__((ext_vector_type(8)))  float    v8f;
typedef __attribute__((ext_vector_type(4)))  int      v4i;

#define BM 128
#define BN 256
#define BK 32
#define NTHREADS 256

// gfx1250 async global->LDS path (ASYNCcnt), with sync fallback.
#if defined(__gfx1250__) && __has_builtin(__builtin_amdgcn_global_load_async_to_lds_b128)
#define USE_ASYNC_LDS 1
typedef __attribute__((address_space(1))) v4i g_v4i;   // global
typedef __attribute__((address_space(3))) v4i l_v4i;   // LDS
#else
#define USE_ASYNC_LDS 0
#endif

__device__ __forceinline__ void async_copy_b128(const _Float16* src, _Float16* lds_dst) {
#if USE_ASYNC_LDS
  __builtin_amdgcn_global_load_async_to_lds_b128(
      (g_v4i*)src, (l_v4i*)lds_dst, 0, 0);
#else
  *(v8h*)lds_dst = *(const v8h*)src;
#endif
}

__device__ __forceinline__ void wait_async0() {
#if USE_ASYNC_LDS
#if __has_builtin(__builtin_amdgcn_s_wait_asynccnt)
  __builtin_amdgcn_s_wait_asynccnt(0);
#else
  asm volatile("s_wait_asynccnt 0" ::: "memory");
#endif
#endif
}

// ---------------------------------------------------------------------------
// 2:4 soft-threshold helpers: in each group of 4, t = 2nd-smallest |w|,
// out = sign(w) * relu(|w| - t) * scale.
// ---------------------------------------------------------------------------
__device__ __forceinline__ float thr4(float a0, float a1, float a2, float a3) {
  float lo0 = fminf(a0, a1), hi0 = fmaxf(a0, a1);
  float lo1 = fminf(a2, a3), hi1 = fmaxf(a2, a3);
  return fminf(fmaxf(lo0, lo1), fminf(hi0, hi1));  // 2nd smallest of 4
}

__device__ __forceinline__ float soft1(float w, float thr, float s) {
  return copysignf(fmaxf(fabsf(w) - thr, 0.0f), w) * s;
}

// weight_in [D,R] fp32, groups of 4 along R  ->  win_t [R,D] fp16 (N-major)
__global__ __launch_bounds__(NTHREADS)
void prep_win(const float* __restrict__ w, const float* __restrict__ scale_p,
              _Float16* __restrict__ wt, int D, int R) {
  int g = blockIdx.x * blockDim.x + threadIdx.x;
  int rg4 = R >> 2;
  if (g >= D * rg4) return;
  int d  = g / rg4;
  int rg = g - d * rg4;
  const float4 wv = *(const float4*)(w + (size_t)d * R + ((size_t)rg << 2));
  float s = *scale_p;
  float t = thr4(fabsf(wv.x), fabsf(wv.y), fabsf(wv.z), fabsf(wv.w));
  int r = rg << 2;
  wt[(size_t)(r + 0) * D + d] = (_Float16)soft1(wv.x, t, s);
  wt[(size_t)(r + 1) * D + d] = (_Float16)soft1(wv.y, t, s);
  wt[(size_t)(r + 2) * D + d] = (_Float16)soft1(wv.z, t, s);
  wt[(size_t)(r + 3) * D + d] = (_Float16)soft1(wv.w, t, s);
}

// weight_out [D,R] fp32, threshold over weight_out.T's last dim (groups of 4
// consecutive d for fixed r)  ->  wout_t [D,R] fp16 (N-major for GEMM2: N=D,K=R)
__global__ __launch_bounds__(NTHREADS)
void prep_wout(const float* __restrict__ w, const float* __restrict__ scale_p,
               _Float16* __restrict__ wt, int D, int R) {
  int g = blockIdx.x * blockDim.x + threadIdx.x;
  if (g >= (D >> 2) * R) return;
  int dg = g / R;
  int r  = g - dg * R;
  int d  = dg << 2;
  float w0 = w[(size_t)(d + 0) * R + r];
  float w1 = w[(size_t)(d + 1) * R + r];
  float w2 = w[(size_t)(d + 2) * R + r];
  float w3 = w[(size_t)(d + 3) * R + r];
  float s = *scale_p;
  float t = thr4(fabsf(w0), fabsf(w1), fabsf(w2), fabsf(w3));
  wt[(size_t)(d + 0) * R + r] = (_Float16)soft1(w0, t, s);
  wt[(size_t)(d + 1) * R + r] = (_Float16)soft1(w1, t, s);
  wt[(size_t)(d + 2) * R + r] = (_Float16)soft1(w2, t, s);
  wt[(size_t)(d + 3) * R + r] = (_Float16)soft1(w3, t, s);
}

// ---------------------------------------------------------------------------
// WMMA GEMM: C[M,N] = A[M,K] * B[K,N] (+bias), B supplied N-major (Bt[N,K] f16).
// 256 threads = 8 waves; block tile 128x256, BK=32 (one v_wmma K-step).
// Wave grid 2(M) x 4(N); each wave: 4x4 fragments of 16x16 -> 16 WMMA/K-step.
// ---------------------------------------------------------------------------
template <typename TA, typename TC, bool HAS_BIAS>
__global__ __launch_bounds__(NTHREADS)
void gemm_wmma(const TA* __restrict__ A, const _Float16* __restrict__ Bt,
               TC* __restrict__ C, const float* __restrict__ bias,
               int M, int N, int K) {
  __shared__ _Float16 As[BM][BK + 8];   // +16B pad: keeps 16B align, dodges bank conflicts
  __shared__ _Float16 Bs[BN][BK + 8];

  const int tid   = threadIdx.x;
  const int lane  = tid & 31;
  const int wid   = tid >> 5;
  const int waveM = wid & 1;    // 64-row slice of the 128-row tile
  const int waveN = wid >> 1;   // 64-col slice of the 256-col tile
  const int m0 = blockIdx.x * BM;
  const int n0 = blockIdx.y * BN;
  const int lr = lane & 15;     // row (A) / col (B) within fragment
  const int lk = lane >> 4;     // K-half select per ISA 7.12.2 layouts

  v8f acc[4][4] = {};

  for (int kt = 0; kt < K; kt += BK) {
    __syncthreads();
    // B tile (BN x BK f16): async global->LDS, 4 x b128 chunks per thread.
    #pragma unroll
    for (int i = 0; i < 4; ++i) {
      int c   = tid + i * NTHREADS;   // 1024 chunks of 8 f16
      int row = c >> 2;               // 0..255
      int col = (c & 3) << 3;         // 0,8,16,24
      async_copy_b128(Bt + (size_t)(n0 + row) * K + kt + col, &Bs[row][col]);
    }
    // A tile (BM x BK): through VGPRs (needs fp32->fp16 convert when TA==float).
    #pragma unroll
    for (int i = 0; i < 2; ++i) {
      int c   = tid + i * NTHREADS;   // 512 chunks of 8 f16
      int row = c >> 2;               // 0..127
      int col = (c & 3) << 3;         // 0,8,16,24
      const TA* srcA = A + (size_t)(m0 + row) * K + kt + col;
      v8h av;
      if constexpr (sizeof(TA) == 4) {
        const float4 f0 = *(const float4*)(srcA);
        const float4 f1 = *(const float4*)(srcA + 4);
        av[0] = (_Float16)f0.x; av[1] = (_Float16)f0.y;
        av[2] = (_Float16)f0.z; av[3] = (_Float16)f0.w;
        av[4] = (_Float16)f1.x; av[5] = (_Float16)f1.y;
        av[6] = (_Float16)f1.z; av[7] = (_Float16)f1.w;
      } else {
        av = *(const v8h*)srcA;
      }
      *(v8h*)&As[row][col] = av;
    }
    // Prefetch next K-tile into L2 (global_prefetch_b8).
    if (kt + BK < K) {
      __builtin_prefetch(A  + (size_t)(m0 + (tid >> 1)) * K + kt + BK, 0, 1);
      __builtin_prefetch(Bt + (size_t)(n0 + tid) * K + kt + BK, 0, 1);
    }
    wait_async0();
    __syncthreads();

    // A fragments: lane lr = M-row; lk picks K[0:8)+K[16:24) vs K[8:16)+K[24:32)
    v16h af[4];
    #pragma unroll
    for (int mt = 0; mt < 4; ++mt) {
      const _Float16* ap = &As[waveM * 64 + mt * 16 + lr][lk * 8];
      *((v8h*)&af[mt] + 0) = *(const v8h*)(ap);
      *((v8h*)&af[mt] + 1) = *(const v8h*)(ap + 16);
    }
    // B fragments: lane lr = N-col; lk picks K[0:8)+K[16:24) vs K[8:16)+K[24:32)
    v16h bf[4];
    #pragma unroll
    for (int nt = 0; nt < 4; ++nt) {
      const _Float16* bp = &Bs[waveN * 64 + nt * 16 + lr][lk * 8];
      *((v8h*)&bf[nt] + 0) = *(const v8h*)(bp);
      *((v8h*)&bf[nt] + 1) = *(const v8h*)(bp + 16);
    }
    #pragma unroll
    for (int mt = 0; mt < 4; ++mt)
      #pragma unroll
      for (int nt = 0; nt < 4; ++nt)
        acc[mt][nt] = __builtin_amdgcn_wmma_f32_16x16x32_f16(
            false, af[mt], false, bf[nt], (short)0, acc[mt][nt], false, false);
  }

  // Epilogue. C/D layout: VGPR r, lanes 0-15 -> M=r, lanes 16-31 -> M=r+8.
  #pragma unroll
  for (int mt = 0; mt < 4; ++mt) {
    int mbase = m0 + waveM * 64 + mt * 16 + lk * 8;
    #pragma unroll
    for (int nt = 0; nt < 4; ++nt) {
      int col = n0 + waveN * 64 + nt * 16 + lr;
      float bv = 0.0f;
      if constexpr (HAS_BIAS) bv = bias[col];
      #pragma unroll
      for (int r = 0; r < 8; ++r) {
        float v = acc[mt][nt][r] + bv;
        C[(size_t)(mbase + r) * N + col] = (TC)v;
      }
    }
  }
}

// ---------------------------------------------------------------------------
// Launcher
// ---------------------------------------------------------------------------
extern "C" void kernel_launch(void* const* d_in, const int* in_sizes, int n_in,
                              void* d_out, int out_size, void* d_ws, size_t ws_size,
                              hipStream_t stream) {
  (void)in_sizes; (void)n_in; (void)out_size; (void)ws_size;
  const int D = 2048, R = 512, M = 4 * 2048;  // B*S

  const float* x      = (const float*)d_in[0];
  const float* w_in   = (const float*)d_in[1];
  const float* w_out  = (const float*)d_in[2];
  const float* bias   = (const float*)d_in[3];
  const float* sc_in  = (const float*)d_in[4];
  const float* sc_out = (const float*)d_in[5];
  float* out = (float*)d_out;

  // Workspace: win_t [R,D] f16 (2MB) | wout_t [D,R] f16 (2MB) | h [M,R] f16 (8MB)
  _Float16* win_t  = (_Float16*)d_ws;
  _Float16* wout_t = win_t + (size_t)R * D;
  _Float16* h      = wout_t + (size_t)D * R;

  int ng1 = D * (R / 4);
  prep_win<<<(ng1 + NTHREADS - 1) / NTHREADS, NTHREADS, 0, stream>>>(
      w_in, sc_in, win_t, D, R);
  int ng2 = (D / 4) * R;
  prep_wout<<<(ng2 + NTHREADS - 1) / NTHREADS, NTHREADS, 0, stream>>>(
      w_out, sc_out, wout_t, D, R);

  // Stage 1: h = f16(x) @ win_s   (M=8192, N=512, K=2048), h stored f16
  gemm_wmma<float, _Float16, false>
      <<<dim3(M / BM, R / BN), NTHREADS, 0, stream>>>(x, win_t, h, nullptr, M, R, D);

  // Stage 2: out = h @ wout_s + bias  (M=8192, N=2048, K=512), f32 out
  gemm_wmma<_Float16, float, true>
      <<<dim3(M / BM, D / BN), NTHREADS, 0, stream>>>(h, wout_t, out, bias, M, D, R);
}
